// RGCN_24661702214223
// MI455X (gfx1250) — matrix operand order repaired and checked by reading, not compiled, not run on hardware
//
#include <hip/hip_runtime.h>
#include <stdint.h>

// ---------------------------------------------------------------------------
// RGCN (2-layer, 8 relations) for MI455X / gfx1250: wave32, WMMA bf16, TDM.
//
// Math rewrite: mean-aggregation is linear, so per-(dst,rel) means are
// computed on RAW features first, then transformed:
//   sum_r ( mean_{e: etype=r, dst=n} x[src_e] ) @ W_r
// Node features live as a 29MB bf16 table -> L2-resident random gathers.
// Big GEMMs (K=128, bf16 A) stage their A tile with one tensor_load_to_lds
// (TDM pads rows in hardware to the bank-conflict-free LDS stride).
// ---------------------------------------------------------------------------

typedef __attribute__((ext_vector_type(16))) __bf16 v16bf;
typedef __attribute__((ext_vector_type(8)))  float  v8f;
typedef __attribute__((ext_vector_type(4)))  unsigned int u32x4;
typedef __attribute__((ext_vector_type(8)))  int          i32x8;

#define NN_USER 50000
#define NN_FOOD 30000
#define NN_ING  20000
#define NN_CAT   5000
#define NN_HAB  10000
#define NN_ALL 115000
#define N_REL 8
#define N_EDGES 1500000
#define DCOM 128

__device__ __forceinline__ unsigned short f2bf(float f) {
    union { float f; unsigned u; } v; v.f = f;
    unsigned r = v.u + 0x7FFFu + ((v.u >> 16) & 1u);   // round-to-nearest-even
    return (unsigned short)(r >> 16);
}
__device__ __forceinline__ float bf2f(unsigned short h) {
    union { unsigned u; float f; } v; v.u = ((unsigned)h) << 16;
    return v.f;
}

// ===========================================================================
// GEMM A (f32 source, generic K multiple of 32): C_bf16[M,128] = A*B + bias
// Used for the node-type projections (K = 256/512/128/64).
// ===========================================================================
#define LDSS 40   // padded row stride (bf16 elems): 80B rows, 16B aligned

__global__ __launch_bounds__(256)
void gemm128_f32(const float* __restrict__ A, const float* __restrict__ B,
                 const float* __restrict__ bias, unsigned short* __restrict__ C,
                 int M, int K)
{
    __shared__ alignas(16) unsigned short As[128 * LDSS];
    __shared__ alignas(16) unsigned short Bs[128 * LDSS];

    const int t        = threadIdx.x;
    const int wave     = t >> 5;
    const int lane     = t & 31;
    const int blockRow = blockIdx.x * 128;

    v8f acc[8];
#pragma unroll
    for (int c = 0; c < 8; ++c)
#pragma unroll
        for (int i = 0; i < 8; ++i) acc[c][i] = 0.0f;

    for (int k0 = 0; k0 < K; k0 += 32) {
        {   // stage A tile 128x32 (f32 -> bf16)
            const int row  = t >> 1;
            const int kb   = (t & 1) * 16;
            const int grow = blockRow + row;
            unsigned short* dstp = &As[row * LDSS + kb];
            if (grow < M) {
                const float4* s4 = (const float4*)(A + (size_t)grow * K + k0 + kb);
#pragma unroll
                for (int q = 0; q < 4; ++q) {
                    float4 v = s4[q];
                    dstp[q * 4 + 0] = f2bf(v.x);
                    dstp[q * 4 + 1] = f2bf(v.y);
                    dstp[q * 4 + 2] = f2bf(v.z);
                    dstp[q * 4 + 3] = f2bf(v.w);
                }
            } else {
#pragma unroll
                for (int q = 0; q < 16; ++q) dstp[q] = 0;
            }
        }
        {   // stage B tile transposed: Bs[col][k]
            const int col = t & 127;
            const int kb  = (t >> 7) * 16;
            const float* Bp = B + (size_t)(k0 + kb) * DCOM + col;
            unsigned short* bd = &Bs[col * LDSS + kb];
#pragma unroll
            for (int j = 0; j < 16; ++j) bd[j] = f2bf(Bp[(size_t)j * DCOM]);
        }
        __syncthreads();

        const int arow  = wave * 16 + (lane & 15);
        const int koffA = (lane < 16) ? 0 : 8;
        union { uint4 q[2]; v16bf v; } au;
        au.q[0] = *(const uint4*)&As[arow * LDSS + koffA];
        au.q[1] = *(const uint4*)&As[arow * LDSS + 16 + koffA];
        const int koffB = (lane < 16) ? 0 : 8;
#pragma unroll
        for (int c = 0; c < 8; ++c) {
            const int bcol = c * 16 + (lane & 15);
            union { uint4 q[2]; v16bf v; } bu;
            bu.q[0] = *(const uint4*)&Bs[bcol * LDSS + koffB];
            bu.q[1] = *(const uint4*)&Bs[bcol * LDSS + koffB + 16];
            acc[c] = __builtin_amdgcn_wmma_f32_16x16x32_bf16(
                false, au.v, false, bu.v, (short)0, acc[c], false, false);
        }
        __syncthreads();
    }

    const int rbase = blockRow + wave * 16 + ((lane >> 4) << 3);
    const int nlo   = lane & 15;
#pragma unroll
    for (int c = 0; c < 8; ++c) {
        const int colg = c * 16 + nlo;
        const float badd = bias[colg];
#pragma unroll
        for (int i = 0; i < 8; ++i) {
            const int gr = rbase + i;
            if (gr < M) C[(size_t)gr * DCOM + colg] = f2bf(acc[c][i] + badd);
        }
    }
}

// ===========================================================================
// GEMM B (bf16 source, K=128 fixed): TDM-staged A tile, whole-B resident.
//   OUT_MODE 0: Cf = acc + bias (f32)    OUT_MODE 2: Cf += acc (f32)
// One tensor_load_to_lds moves the 128x128 bf16 A tile into LDS with
// hardware row padding (pad_interval=64 DW -> every 256B, pad_amount=4 DW);
// resulting LDS row stride = 272B = 136 bf16 elems. OOB rows (last block)
// read as zeros via tensor_dim1 clamping.
// ===========================================================================
#define LDST 136

template <int OUT_MODE>
__global__ __launch_bounds__(256)
void gemm128_tdm(const unsigned short* __restrict__ Abf, const float* __restrict__ B,
                 const float* __restrict__ bias, float* __restrict__ Cf, int M)
{
    __shared__ alignas(16) unsigned short As[128 * LDST];  // 34816 B
    __shared__ alignas(16) unsigned short Bs[128 * LDST];  // 34816 B

    const int t        = threadIdx.x;
    const int wave     = t >> 5;
    const int lane     = t & 31;
    const int blockRow = blockIdx.x * 128;

    // ---- wave 0: issue TDM DMA of the A tile ------------------------------
    if (t < 32) {
        const unsigned ldsb = (unsigned)(uintptr_t)(&As[0]);      // LDS byte offset
        const uint64_t ga   = (uint64_t)(uintptr_t)(Abf + (size_t)blockRow * DCOM);
        const unsigned td1  = (unsigned)(M - blockRow);           // rows remaining
        u32x4 g0;
        g0.x = 1u;                                                // count=1, user D#
        g0.y = ldsb;                                              // lds_addr
        g0.z = (unsigned)ga;                                      // global_addr[31:0]
        g0.w = (unsigned)((ga >> 32) & 0x1FFFFFFull) | (2u << 30); // addr[56:32]|type=2
        i32x8 g1;
        g1[0] = (int)((1u << 16)      // data_size = 2 bytes
                    | (1u << 20)      // pad_enable
                    | (5u << 22)      // pad_interval: 64 DWORDs (256B = one row)
                    | (3u << 25));    // pad_amount:   4 DWORDs (16B)
        g1[1] = (int)(128u << 16);                                 // tensor_dim0 = 128 (lo16)
        g1[2] = (int)((td1 & 0xFFFFu) << 16);                      // dim0 hi=0 | tensor_dim1 lo16
        g1[3] = (int)(((td1 >> 16) & 0xFFFFu) | (128u << 16));     // dim1 hi | tile_dim0 = 128
        g1[4] = (int)128u;                                         // tile_dim1 = 128, tile_dim2 = 0
        g1[5] = (int)128u;                                         // tensor_dim0_stride = 128 (lo32)
        g1[6] = 0;                                                 // stride hi | dim1_stride lo (unused, 2D)
        g1[7] = 0;
        asm volatile("tensor_load_to_lds %0, %1" :: "s"(g0), "s"(g1) : "memory");
    }

    // ---- all waves: stage full B (f32 -> bf16, transposed [col][k]) -------
    {
        const int col = t & 127;
        const int kb  = (t >> 7) * 64;
        unsigned short* bd = &Bs[col * LDST + kb];
        const float* Bp = B + (size_t)kb * DCOM + col;
#pragma unroll 8
        for (int j = 0; j < 64; ++j) bd[j] = f2bf(Bp[(size_t)j * DCOM]);
    }

    if (t < 32) __builtin_amdgcn_s_wait_tensorcnt(0);   // DMA complete (wave 0)
    __syncthreads();                                    // publish LDS to all waves

    // ---- compute: 4 K-steps x 8 col tiles, no further barriers ------------
    v8f acc[8];
#pragma unroll
    for (int c = 0; c < 8; ++c)
#pragma unroll
        for (int i = 0; i < 8; ++i) acc[c][i] = 0.0f;

    const int arow  = wave * 16 + (lane & 15);
    const int koffA = (lane < 16) ? 0 : 8;
    const int koffB = (lane < 16) ? 0 : 8;
    const int nlo   = lane & 15;

#pragma unroll
    for (int k0 = 0; k0 < 128; k0 += 32) {
        union { uint4 q[2]; v16bf v; } au;
        au.q[0] = *(const uint4*)&As[arow * LDST + k0 + koffA];
        au.q[1] = *(const uint4*)&As[arow * LDST + k0 + 16 + koffA];
#pragma unroll
        for (int c = 0; c < 8; ++c) {
            const int bcol = c * 16 + nlo;
            union { uint4 q[2]; v16bf v; } bu;
            bu.q[0] = *(const uint4*)&Bs[bcol * LDST + k0 + koffB];
            bu.q[1] = *(const uint4*)&Bs[bcol * LDST + k0 + koffB + 16];
            acc[c] = __builtin_amdgcn_wmma_f32_16x16x32_bf16(
                false, au.v, false, bu.v, (short)0, acc[c], false, false);
        }
    }

    // ---- epilogue ---------------------------------------------------------
    const int rbase = blockRow + wave * 16 + ((lane >> 4) << 3);
#pragma unroll
    for (int c = 0; c < 8; ++c) {
        const int colg = c * 16 + nlo;
        float badd = 0.0f;
        if (OUT_MODE == 0 && bias != nullptr) badd = bias[colg];
#pragma unroll
        for (int i = 0; i < 8; ++i) {
            const int gr = rbase + i;
            if (gr < M) {
                const size_t idx = (size_t)gr * DCOM + colg;
                if (OUT_MODE == 0) Cf[idx] = acc[c][i] + badd;
                else               Cf[idx] += acc[c][i];
            }
        }
    }
}

// ---------------------------------------------------------------------------
// Edge / elementwise kernels
// ---------------------------------------------------------------------------
__global__ void zero_f32(float* __restrict__ p, int n) {
    int i = blockIdx.x * blockDim.x + threadIdx.x;
    if (i < n) p[i] = 0.0f;
}

__global__ void count_edges(const int* __restrict__ dst, const int* __restrict__ et,
                            float* __restrict__ cnt, int E) {
    int e = blockIdx.x * blockDim.x + threadIdx.x;
    if (e < E) atomicAdd(&cnt[(size_t)dst[e] * N_REL + et[e]], 1.0f);
}

// one wave per edge; lane covers 4 of 128 features; bf16 gather (L2-resident)
__global__ void scatter_rel(const int* __restrict__ src, const int* __restrict__ dst,
                            const int* __restrict__ et,
                            const unsigned short* __restrict__ x,
                            float* __restrict__ sums, int E, int r) {
    int gid  = blockIdx.x * blockDim.x + threadIdx.x;
    int e    = gid >> 5;
    int lane = gid & 31;
    if (e >= E) return;
    if (et[e] != r) return;
    int s = src[e], d = dst[e];
    uint2 q = *(const uint2*)(x + (size_t)s * DCOM + lane * 4);
    float* o = sums + (size_t)d * DCOM + lane * 4;
    atomicAdd(o + 0, bf2f((unsigned short)(q.x & 0xFFFFu)));
    atomicAdd(o + 1, bf2f((unsigned short)(q.x >> 16)));
    atomicAdd(o + 2, bf2f((unsigned short)(q.y & 0xFFFFu)));
    atomicAdd(o + 3, bf2f((unsigned short)(q.y >> 16)));
}

// mean + bf16 convert; resets sums to 0 for the next relation pass
__global__ void mean_to_bf16(float* __restrict__ sums, const float* __restrict__ cnt,
                             unsigned short* __restrict__ mb, int N, int r) {
    int i = blockIdx.x * blockDim.x + threadIdx.x;
    if (i >= N * DCOM) return;
    int n = i >> 7;
    float c = cnt[(size_t)n * N_REL + r];
    mb[i] = f2bf(sums[i] / fmaxf(c, 1.0f));
    sums[i] = 0.0f;
}

__global__ void relu_to_bf16(const float* __restrict__ in, unsigned short* __restrict__ out,
                             int total) {
    int i = blockIdx.x * blockDim.x + threadIdx.x;
    if (i < total) out[i] = f2bf(fmaxf(in[i], 0.0f));
}

// layer-2 (Dout=2) root: logits = x2 @ root2 + b2   (trivial FLOPs -> VALU)
__global__ void l2_root(const unsigned short* __restrict__ x2, const float* __restrict__ root2,
                        const float* __restrict__ b2, float* __restrict__ out, int N) {
    int n = blockIdx.x * blockDim.x + threadIdx.x;
    if (n >= N) return;
    float a0 = b2[0], a1 = b2[1];
    const unsigned short* xp = x2 + (size_t)n * DCOM;
#pragma unroll 4
    for (int k = 0; k < DCOM; ++k) {
        float v = bf2f(xp[k]);
        a0 += v * root2[k * 2 + 0];
        a1 += v * root2[k * 2 + 1];
    }
    out[n * 2 + 0] = a0;
    out[n * 2 + 1] = a1;
}

// layer-2 per-relation: logits += mean_r @ W2[r]; resets sums to 0
__global__ void l2_rel(float* __restrict__ sums, const float* __restrict__ cnt,
                       const float* __restrict__ W2r, float* __restrict__ out, int N, int r) {
    int n = blockIdx.x * blockDim.x + threadIdx.x;
    if (n >= N) return;
    float inv = 1.0f / fmaxf(cnt[(size_t)n * N_REL + r], 1.0f);
    float a0 = 0.0f, a1 = 0.0f;
    float* sp = sums + (size_t)n * DCOM;
#pragma unroll 4
    for (int k = 0; k < DCOM; ++k) {
        float v = sp[k] * inv;
        a0 += v * W2r[k * 2 + 0];
        a1 += v * W2r[k * 2 + 1];
        sp[k] = 0.0f;
    }
    out[n * 2 + 0] += a0;
    out[n * 2 + 1] += a1;
}

__global__ void logsoftmax2(float* __restrict__ out, int N) {
    int n = blockIdx.x * blockDim.x + threadIdx.x;
    if (n >= N) return;
    float a = out[n * 2], b = out[n * 2 + 1];
    float m = fmaxf(a, b);
    float lse = m + logf(expf(a - m) + expf(b - m));
    out[n * 2 + 0] = a - lse;
    out[n * 2 + 1] = b - lse;
}

// ---------------------------------------------------------------------------
extern "C" void kernel_launch(void* const* d_in, const int* in_sizes, int n_in,
                              void* d_out, int out_size, void* d_ws, size_t ws_size,
                              hipStream_t stream) {
    const float* x_user = (const float*)d_in[0];
    const float* x_food = (const float*)d_in[1];
    const float* x_ing  = (const float*)d_in[2];
    const float* x_cat  = (const float*)d_in[3];
    const float* x_hab  = (const float*)d_in[4];
    const float* Wu = (const float*)d_in[5];  const float* bu = (const float*)d_in[6];
    const float* Wf = (const float*)d_in[7];  const float* bf = (const float*)d_in[8];
    const float* Wi = (const float*)d_in[9];  const float* bi = (const float*)d_in[10];
    const float* Wc = (const float*)d_in[11]; const float* bc = (const float*)d_in[12];
    const float* Wh = (const float*)d_in[13]; const float* bh = (const float*)d_in[14];
    const float* W1    = (const float*)d_in[15];
    const float* root1 = (const float*)d_in[16];
    const float* b1    = (const float*)d_in[17];
    const float* W2    = (const float*)d_in[18];
    const float* root2 = (const float*)d_in[19];
    const float* b2    = (const float*)d_in[20];
    const int* edge_index = (const int*)d_in[21];
    const int* etype      = (const int*)d_in[22];
    const int* esrc = edge_index;
    const int* edst = edge_index + N_EDGES;
    float* logits = (float*)d_out;

    // ---- workspace layout (all offsets multiple of 256B) ------------------
    char* ws = (char*)d_ws;
    size_t o = 0;
    unsigned short* xall  = (unsigned short*)(ws + o); o += (size_t)NN_ALL * DCOM * 2; // 29.44MB
    float*          sums  = (float*)(ws + o);          o += (size_t)NN_ALL * DCOM * 4; // 58.88MB
    unsigned short* meanb = (unsigned short*)(ws + o); o += (size_t)NN_ALL * DCOM * 2; // 29.44MB
    float*          cnt   = (float*)(ws + o);          o += (size_t)NN_ALL * N_REL * 4;// 3.68MB
    float*          out1  = (float*)(ws + o);          o += (size_t)NN_ALL * DCOM * 4; // 58.88MB
    unsigned short* x2    = (unsigned short*)(ws + o); o += (size_t)NN_ALL * DCOM * 2; // 29.44MB
    // total ~210MB (<= ws_size assumed)

    const dim3 blk(256);
    auto gblk = [](int M) { return dim3((unsigned)((M + 127) / 128)); };
    const int NF = NN_ALL * DCOM;

    // 1) per-type projections -> xall (bf16, bias fused)
    gemm128_f32<<<gblk(NN_USER), blk, 0, stream>>>(x_user, Wu, bu, xall, NN_USER, 256);
    gemm128_f32<<<gblk(NN_FOOD), blk, 0, stream>>>(x_food, Wf, bf, xall + (size_t)NN_USER * DCOM, NN_FOOD, 512);
    gemm128_f32<<<gblk(NN_ING),  blk, 0, stream>>>(x_ing,  Wi, bi, xall + (size_t)(NN_USER + NN_FOOD) * DCOM, NN_ING, 128);
    gemm128_f32<<<gblk(NN_CAT),  blk, 0, stream>>>(x_cat,  Wc, bc, xall + (size_t)(NN_USER + NN_FOOD + NN_ING) * DCOM, NN_CAT, 64);
    gemm128_f32<<<gblk(NN_HAB),  blk, 0, stream>>>(x_hab,  Wh, bh, xall + (size_t)(NN_USER + NN_FOOD + NN_ING + NN_CAT) * DCOM, NN_HAB, 64);

    // 2) deterministic workspace init (ws is poisoned before timing)
    zero_f32<<<(NF + 255) / 256, blk, 0, stream>>>(sums, NF);
    zero_f32<<<(NN_ALL * N_REL + 255) / 256, blk, 0, stream>>>(cnt, NN_ALL * N_REL);
    count_edges<<<(N_EDGES + 255) / 256, blk, 0, stream>>>(edst, etype, cnt, N_EDGES);

    // 3) layer 1 root: out1 = xall @ root1 + b1 (f32, TDM-staged GEMM)
    gemm128_tdm<0><<<gblk(NN_ALL), blk, 0, stream>>>(xall, root1, b1, out1, NN_ALL);

    // 4) layer 1 relations: out1 += mean_r @ W1[r]
    for (int r = 0; r < N_REL; ++r) {
        scatter_rel<<<(int)(((size_t)N_EDGES * 32 + 255) / 256), blk, 0, stream>>>(
            esrc, edst, etype, xall, sums, N_EDGES, r);
        mean_to_bf16<<<(NF + 255) / 256, blk, 0, stream>>>(sums, cnt, meanb, NN_ALL, r);
        gemm128_tdm<2><<<gblk(NN_ALL), blk, 0, stream>>>(
            meanb, W1 + (size_t)r * DCOM * DCOM, nullptr, out1, NN_ALL);
    }

    // 5) ReLU -> bf16 features for layer 2
    relu_to_bf16<<<(NF + 255) / 256, blk, 0, stream>>>(out1, x2, NF);

    // 6) layer 2 root (Dout=2, trivial FLOPs)
    l2_root<<<(NN_ALL + 255) / 256, blk, 0, stream>>>(x2, root2, b2, logits, NN_ALL);

    // 7) layer 2 relations
    for (int r = 0; r < N_REL; ++r) {
        scatter_rel<<<(int)(((size_t)N_EDGES * 32 + 255) / 256), blk, 0, stream>>>(
            esrc, edst, etype, x2, sums, N_EDGES, r);
        l2_rel<<<(NN_ALL + 255) / 256, blk, 0, stream>>>(
            sums, cnt, W2 + (size_t)r * DCOM * 2, logits, NN_ALL, r);
    }

    // 8) log_softmax over 2 classes
    logsoftmax2<<<(NN_ALL + 255) / 256, blk, 0, stream>>>(logits, NN_ALL);
}
